// LSTMblock_3908420239873
// MI455X (gfx1250) — matrix operand order repaired and checked by reading, not compiled, not run on hardware
//
#include <hip/hip_runtime.h>

#define B_   64
#define T_   1024
#define I_   512
#define H_   512
#define L_   2
#define KTOT (H_ + I_)   // 1024

typedef __attribute__((ext_vector_type(16))) __bf16 v16bf;
typedef __attribute__((ext_vector_type(8)))  float  v8f;

static __device__ __forceinline__ unsigned short f2bf(float f) {
    unsigned u = __float_as_uint(f);
    u += 0x7FFFu + ((u >> 16) & 1u);       // round-to-nearest-even
    return (unsigned short)(u >> 16);
}

union FragB16 { uint4 q[2]; v16bf v; };

// ---- pack W{f,i,c,o}[l][k][n] (f32) -> WzT[l][g][n][k] (bf16, k-major) ----
__global__ __launch_bounds__(256) void pack_weights(
    const float* __restrict__ Wf, const float* __restrict__ Wi,
    const float* __restrict__ Wc, const float* __restrict__ Wo,
    unsigned short* __restrict__ WzT)
{
    int e = blockIdx.x * blockDim.x + threadIdx.x;      // over L_*H_*KTOT
    if (e >= L_ * H_ * KTOT) return;
    int k = e % KTOT;
    int n = (e / KTOT) % H_;
    int l = e / (KTOT * H_);
    int src = (l * KTOT + k) * H_ + n;
    const float* Ws[4] = { Wf, Wi, Wc, Wo };
#pragma unroll
    for (int g = 0; g < 4; ++g)
        WzT[((size_t)((l * 4 + g) * H_) + n) * KTOT + k] = f2bf(Ws[g][src]);
}

// ---- x (f32 [B,T,I]) -> bf16 same layout ----
__global__ __launch_bounds__(256) void pack_x(const float* __restrict__ x,
                                              unsigned short* __restrict__ Xbf)
{
    int e = blockIdx.x * blockDim.x + threadIdx.x;
    if (e >= B_ * T_ * I_) return;
    Xbf[e] = f2bf(x[e]);
}

// ---- zero h/c state (f32) and h (bf16) ----
__global__ __launch_bounds__(256) void init_state(float* __restrict__ h_state,
                                                  float* __restrict__ c_state,
                                                  unsigned short* __restrict__ h_bf16)
{
    int e = blockIdx.x * blockDim.x + threadIdx.x;
    if (e >= L_ * B_ * H_) return;
    h_state[e] = 0.f;
    c_state[e] = 0.f;
    h_bf16[e]  = 0;
}

// ---- one recurrent step: both layers, all 4 gates, fused GEMM + gating ----
__global__ __launch_bounds__(256) void lstm_step(
    const unsigned short* __restrict__ Xbf,   // [B][T][I] bf16
    const unsigned short* __restrict__ WzT,   // [L][4][H][KTOT] bf16
    const float* __restrict__ bf_, const float* __restrict__ bi_,
    const float* __restrict__ bc_, const float* __restrict__ bo_,
    float* __restrict__ h_state, float* __restrict__ c_state,
    unsigned short* __restrict__ h_bf16,
    float* __restrict__ out, int t)
{
    const int wave = (blockIdx.x * blockDim.x + threadIdx.x) >> 5;  // 0..255
    const int lane = threadIdx.x & 31;
    const int l    = wave >> 7;            // layer 0..1
    const int rem  = wave & 127;
    const int m0   = (rem >> 5) << 4;      // batch tile origin (0,16,32,48)
    const int n0   = (rem & 31) << 4;      // output-col tile origin
    const int half = lane >> 4;
    const int lm   = lane & 15;
    const int brow_b = m0 + lm;            // batch row this lane loads A from
    const int n      = n0 + lm;            // output column (B/C/D layout)

    // accumulators seeded with bias (bias depends only on N = same for all 8 rows)
    v8f accf, acci, accg, acco;
    const float vbf = bf_[l * H_ + n], vbi = bi_[l * H_ + n];
    const float vbc = bc_[l * H_ + n], vbo = bo_[l * H_ + n];
#pragma unroll
    for (int j = 0; j < 8; ++j) { accf[j] = vbf; acci[j] = vbi; accg[j] = vbc; acco[j] = vbo; }

    const unsigned short* arow_h = h_bf16 + (size_t)(l * B_ + brow_b) * H_;
    const unsigned short* arow_x = Xbf + ((size_t)brow_b * T_ + t) * I_;
    const unsigned short* brow[4];
#pragma unroll
    for (int g = 0; g < 4; ++g)
        brow[g] = WzT + ((size_t)((l * 4 + g) * H_) + n) * KTOT;

    // pull next step's x row toward the WGP while we compute
    if (t + 1 < T_) __builtin_prefetch(arow_x + I_, 0, 1);

    for (int kt = 0; kt < KTOT / 32; ++kt) {
        const int k0 = kt * 32;   // wave-uniform
        // A fragment: ISA 16-bit A 16x32 layout -> two b128 loads per lane
        const unsigned short* arow = (k0 < H_) ? (arow_h + k0) : (arow_x + (k0 - H_));
        FragB16 a;
        const uint4* ap = (const uint4*)arow;
        a.q[0] = ap[half];       // bytes k0*2 + half*16
        a.q[1] = ap[2 + half];   // bytes k0*2 + 32 + half*16
        // B fragments: k-major pre-transposed weights -> two b128 loads per lane, same k order as A
        FragB16 bw[4];
#pragma unroll
        for (int g = 0; g < 4; ++g) {
            const uint4* bp = (const uint4*)(brow[g] + k0);
            bw[g].q[0] = bp[half];
            bw[g].q[1] = bp[2 + half];
        }
        // 4 gate WMMAs reuse the A fragment
        accf = __builtin_amdgcn_wmma_f32_16x16x32_bf16(false, a.v, false, bw[0].v, (short)0, accf, false, false);
        acci = __builtin_amdgcn_wmma_f32_16x16x32_bf16(false, a.v, false, bw[1].v, (short)0, acci, false, false);
        accg = __builtin_amdgcn_wmma_f32_16x16x32_bf16(false, a.v, false, bw[2].v, (short)0, accg, false, false);
        acco = __builtin_amdgcn_wmma_f32_16x16x32_bf16(false, a.v, false, bw[3].v, (short)0, acco, false, false);
    }

    // elementwise LSTM update per C/D f32 layout: VGPR j -> M = half*8 + j, N = lane&15
#pragma unroll
    for (int j = 0; j < 8; ++j) {
        const int b   = m0 + half * 8 + j;
        const size_t idx = (size_t)(l * B_ + b) * H_ + n;
        const float fg = 1.f / (1.f + __expf(-accf[j]));
        const float ig = 1.f / (1.f + __expf(-acci[j]));
        const float gg = tanhf(accg[j]);
        const float og = 1.f / (1.f + __expf(-acco[j]));
        const float c  = fg * c_state[idx] + ig * gg;
        const float h  = og * tanhf(c);
        c_state[idx] = c;
        h_state[idx] = h;
        h_bf16[idx]  = f2bf(h);
        if (l == L_ - 1) out[((size_t)b * T_ + t) * H_ + n] = h;
    }
}

// ---- copy hT, cT into d_out tail ----
__global__ __launch_bounds__(256) void copy_tails(const float* __restrict__ h_state,
                                                  const float* __restrict__ c_state,
                                                  float* __restrict__ out_tail)
{
    int e = blockIdx.x * blockDim.x + threadIdx.x;
    if (e >= L_ * B_ * H_) return;
    out_tail[e]                 = h_state[e];
    out_tail[L_ * B_ * H_ + e]  = c_state[e];
}

extern "C" void kernel_launch(void* const* d_in, const int* in_sizes, int n_in,
                              void* d_out, int out_size, void* d_ws, size_t ws_size,
                              hipStream_t stream)
{
    const float* x  = (const float*)d_in[0];
    const float* Wf = (const float*)d_in[1];
    const float* bf = (const float*)d_in[2];
    const float* Wi = (const float*)d_in[3];
    const float* bi = (const float*)d_in[4];
    const float* Wc = (const float*)d_in[5];
    const float* bc = (const float*)d_in[6];
    const float* Wo = (const float*)d_in[7];
    const float* bo = (const float*)d_in[8];
    float* out = (float*)d_out;

    char* ws = (char*)d_ws;
    size_t off = 0;
    unsigned short* WzT = (unsigned short*)(ws + off); off += (size_t)L_ * 4 * H_ * KTOT * 2; // 8 MB
    unsigned short* Xbf = (unsigned short*)(ws + off); off += (size_t)B_ * T_ * I_ * 2;       // 64 MB
    float* h_state      = (float*)(ws + off);          off += (size_t)L_ * B_ * H_ * 4;
    float* c_state      = (float*)(ws + off);          off += (size_t)L_ * B_ * H_ * 4;
    unsigned short* h_bf16 = (unsigned short*)(ws + off);

    {
        int nW = L_ * H_ * KTOT;
        pack_weights<<<(nW + 255) / 256, 256, 0, stream>>>(Wf, Wi, Wc, Wo, WzT);
        int nX = B_ * T_ * I_;
        pack_x<<<(nX + 255) / 256, 256, 0, stream>>>(x, Xbf);
        int nS = L_ * B_ * H_;
        init_state<<<(nS + 255) / 256, 256, 0, stream>>>(h_state, c_state, h_bf16);
    }

    // 256 waves per step: 2 layers x 4 M-tiles x 32 N-tiles; 8 waves/block -> 32 blocks
    for (int t = 0; t < T_; ++t) {
        lstm_step<<<32, 256, 0, stream>>>(Xbf, WzT, bf, bi, bc, bo,
                                          h_state, c_state, h_bf16, out, t);
    }

    {
        int nS = L_ * B_ * H_;
        copy_tails<<<(nS + 255) / 256, 256, 0, stream>>>(h_state, c_state,
                                                         out + (size_t)B_ * T_ * H_);
    }
}
